// SwinAttention_37349035606033
// MI455X (gfx1250) — matrix operand order repaired and checked by reading, not compiled, not run on hardware
//
#include <hip/hip_runtime.h>

// ---------------------------------------------------------------------------
// Swin window attention, fused bf16-WMMA implementation for gfx1250 (MI455X).
// B=2048 windows, SEQ=49 (pad->64), D=384, H=12, hd=32.
// ---------------------------------------------------------------------------

#define WIN 7
#define SEQ 49
#define SPAD 64
#define DIM 384
#define NH 12
#define HD 32

typedef __bf16 bf16_t;
typedef __attribute__((ext_vector_type(16))) __bf16 v16bf;
typedef __attribute__((ext_vector_type(8)))  float  v8f;

union FragU { v16bf v; uint4 q[2]; };

__device__ __forceinline__ bf16_t f2bf(float x) {
  union { float f; unsigned u; } a; a.f = x;
  unsigned r = a.u + 0x7FFFu + ((a.u >> 16) & 1u);   // round-to-nearest-even
  union { unsigned short s; bf16_t b; } c;
  c.s = (unsigned short)(r >> 16);
  return c.b;
}

// A-fragment (16x32 bf16, M x K): lane m = lane&15; K chunks [kb..kb+7] and
// [16+kb..16+kb+7], kb = (lane>=16)*8. Source is row-major [stride] bf16.
__device__ __forceinline__ v16bf load_a_frag(const bf16_t* base, int stride, int lane) {
  const int m  = lane & 15;
  const int kb = (lane >> 4) << 3;
  const bf16_t* p = base + (size_t)m * stride + kb;
  FragU f;
  f.q[0] = *(const uint4*)(p);
  f.q[1] = *(const uint4*)(p + 16);
  return f.v;
}

// B-fragment (32x16 bf16, K x N): lane n = lane&15; K chunks [ko..ko+7] and
// [16+ko..16+ko+7], ko = (lane>=16)*8. Source is "column-major B" == row-major [n][k], stride
// elements per n-row (i.e. weight row j holds its K contraction contiguously).
__device__ __forceinline__ v16bf load_b_frag(const bf16_t* base, int stride, int lane) {
  const int n  = lane & 15;
  const int ko = (lane >> 4) << 3;
  const bf16_t* p = base + (size_t)n * stride + ko;
  FragU f;
  f.q[0] = *(const uint4*)(p);
  f.q[1] = *(const uint4*)(p + 16);
  return f.v;
}

__device__ __forceinline__ v8f wmma_bf16(v16bf a, v16bf b, v8f c) {
  return __builtin_amdgcn_wmma_f32_16x16x32_bf16(false, a, false, b, (short)0, c, false, false);
}

// ---------------------------------------------------------------------------
// Kernel 0: convert weights to bf16 in workspace (stay hot in L2 afterwards).
// ---------------------------------------------------------------------------
__global__ void prep_weights(const float* __restrict__ wqkv, const float* __restrict__ wout,
                             bf16_t* __restrict__ wqkv_b, bf16_t* __restrict__ wout_b) {
  int i = blockIdx.x * blockDim.x + threadIdx.x;
  if (i < 3 * DIM * DIM) wqkv_b[i] = f2bf(wqkv[i]);
  if (i < DIM * DIM)     wout_b[i] = f2bf(wout[i]);
}

// ---------------------------------------------------------------------------
// Kernel 1: fused QKV + attention per window. 1 block per window, 6 waves,
// each wave handles heads {w, w+6}. All intermediates live in LDS.
// LDS layout (bf16 elements):
//   hs   : 64 x 392                    (padded hidden, rows 49..63 zeroed)
//   per wave: q 64x32 | k 64x32 | vT 32x72 | p 16x72
// total = (25088 + 6*7552)*2 = 140800 bytes (dynamic).
// ---------------------------------------------------------------------------
#define HS_STRIDE 392
#define VT_STRIDE 72
#define P_STRIDE  72

__global__ void attn_fused(const float* __restrict__ hidden,
                           const float* __restrict__ amask,
                           const bf16_t* __restrict__ wqkv_b,
                           const float* __restrict__ b_qkv,
                           const float* __restrict__ table,
                           bf16_t* __restrict__ ctx) {
  extern __shared__ __align__(16) char smem_raw[];
  bf16_t* hs = (bf16_t*)smem_raw;

  const int b    = blockIdx.x;
  const int w    = threadIdx.x >> 5;
  const int lane = threadIdx.x & 31;
  const int lanelo = lane & 15;
  const int half   = lane >> 4;

  bf16_t* wavebuf = hs + 25088 + (size_t)w * 7552;
  bf16_t* q_lds  = wavebuf;          // [64][32]
  bf16_t* k_lds  = wavebuf + 2048;   // [64][32]
  bf16_t* vt_lds = wavebuf + 4096;   // [32][72]
  bf16_t* p_lds  = wavebuf + 6400;   // [16][72]

  // --- cooperative load: hidden f32 -> bf16 LDS, zero-pad rows 49..63 ------
  const float* hg = hidden + (size_t)b * SEQ * DIM;
  for (int i = threadIdx.x; i < SEQ * DIM; i += blockDim.x) {
    int s = i / DIM, d = i - s * DIM;
    hs[s * HS_STRIDE + d] = f2bf(hg[i]);
  }
  for (int i = threadIdx.x; i < (SPAD - SEQ) * DIM; i += blockDim.x) {
    int s = SEQ + i / DIM, d = i - (s - SEQ) * DIM;
    hs[s * HS_STRIDE + d] = f2bf(0.0f);
  }
  __syncthreads();

  const float scale = 0.17677669529663687f;  // 1/sqrt(32)

  for (int hh = 0; hh < 2; ++hh) {
    const int h = w + hh * 6;
    const bf16_t* wq = wqkv_b + (size_t)(h * HD) * DIM;
    const bf16_t* wk = wqkv_b + (size_t)(DIM + h * HD) * DIM;
    const bf16_t* wv = wqkv_b + (size_t)(2 * DIM + h * HD) * DIM;

    // ---------------- QKV projection for this head (bf16 WMMA) ------------
    for (int mt = 0; mt < 4; ++mt) {
      for (int nt = 0; nt < 2; ++nt) {
        v8f aq = {}, ak = {}, av = {};
        for (int kt = 0; kt < 12; ++kt) {
          v16bf A  = load_a_frag(hs + (size_t)mt * 16 * HS_STRIDE + kt * 32, HS_STRIDE, lane);
          v16bf Bq = load_b_frag(wq + (size_t)nt * 16 * DIM + kt * 32, DIM, lane);
          v16bf Bk = load_b_frag(wk + (size_t)nt * 16 * DIM + kt * 32, DIM, lane);
          v16bf Bv = load_b_frag(wv + (size_t)nt * 16 * DIM + kt * 32, DIM, lane);
          aq = wmma_bf16(A, Bq, aq);
          ak = wmma_bf16(A, Bk, ak);
          av = wmma_bf16(A, Bv, av);
        }
        const int col = nt * 16 + lanelo;
        const float bq = b_qkv[h * HD + col];
        const float bk = b_qkv[DIM + h * HD + col];
        const float bv = b_qkv[2 * DIM + h * HD + col];
#pragma unroll
        for (int r = 0; r < 8; ++r) {
          const int row = mt * 16 + r + half * 8;
          q_lds[row * HD + col]          = f2bf(aq[r] + bq);
          k_lds[row * HD + col]          = f2bf(ak[r] + bk);
          vt_lds[col * VT_STRIDE + row]  = f2bf(av[r] + bv);   // V transposed
        }
      }
    }

    // ---------------- attention: scores -> softmax -> PV ------------------
    for (int mt = 0; mt < 4; ++mt) {
      v16bf Aq = load_a_frag(q_lds + (size_t)mt * 16 * HD, HD, lane);
      v8f sc[4];
#pragma unroll
      for (int nt = 0; nt < 4; ++nt) {
        v16bf Bk = load_b_frag(k_lds + (size_t)nt * 16 * HD, HD, lane);
        v8f z = {};
        sc[nt] = wmma_bf16(Aq, Bk, z);
      }
      // scale + relative-position bias + attention mask + key padding mask
#pragma unroll
      for (int nt = 0; nt < 4; ++nt) {
        const int k = nt * 16 + lanelo;
#pragma unroll
        for (int r = 0; r < 8; ++r) {
          const int qi = mt * 16 + r + half * 8;
          float s = sc[nt][r] * scale;
          if (k >= SEQ) {
            s = -1e30f;
          } else if (qi < SEQ) {
            const int dr = (qi / WIN) - (k / WIN) + (WIN - 1);
            const int dc = (qi % WIN) - (k % WIN) + (WIN - 1);
            s += table[(dr * (2 * WIN - 1) + dc) * NH + h];
            s += amask[(size_t)b * SEQ * SEQ + (size_t)qi * SEQ + k];
          }
          sc[nt][r] = s;
        }
      }
      // row softmax: reduce across the 16-lane half holding the row's columns
#pragma unroll
      for (int r = 0; r < 8; ++r) {
        float m = fmaxf(fmaxf(sc[0][r], sc[1][r]), fmaxf(sc[2][r], sc[3][r]));
        m = fmaxf(m, __shfl_xor(m, 1, 32));
        m = fmaxf(m, __shfl_xor(m, 2, 32));
        m = fmaxf(m, __shfl_xor(m, 4, 32));
        m = fmaxf(m, __shfl_xor(m, 8, 32));
        float e0 = __expf(sc[0][r] - m);
        float e1 = __expf(sc[1][r] - m);
        float e2 = __expf(sc[2][r] - m);
        float e3 = __expf(sc[3][r] - m);
        float ssum = e0 + e1 + e2 + e3;
        ssum += __shfl_xor(ssum, 1, 32);
        ssum += __shfl_xor(ssum, 2, 32);
        ssum += __shfl_xor(ssum, 4, 32);
        ssum += __shfl_xor(ssum, 8, 32);
        const float inv = 1.0f / ssum;
        const int row = r + half * 8;
        p_lds[row * P_STRIDE +  0 + lanelo] = f2bf(e0 * inv);
        p_lds[row * P_STRIDE + 16 + lanelo] = f2bf(e1 * inv);
        p_lds[row * P_STRIDE + 32 + lanelo] = f2bf(e2 * inv);
        p_lds[row * P_STRIDE + 48 + lanelo] = f2bf(e3 * inv);
      }
      // ctx = probs @ V   (K = 64 padded keys, probs for k>=49 are exactly 0)
#pragma unroll
      for (int nt = 0; nt < 2; ++nt) {
        v8f co = {};
#pragma unroll
        for (int kt = 0; kt < 2; ++kt) {
          v16bf Ap = load_a_frag(p_lds + kt * 32, P_STRIDE, lane);
          v16bf Bv = load_b_frag(vt_lds + (size_t)nt * 16 * VT_STRIDE + kt * 32, VT_STRIDE, lane);
          co = wmma_bf16(Ap, Bv, co);
        }
        const int col = nt * 16 + lanelo;
#pragma unroll
        for (int r = 0; r < 8; ++r) {
          const int qi = mt * 16 + r + half * 8;
          if (qi < SEQ)
            ctx[(size_t)b * SEQ * DIM + (size_t)qi * DIM + h * HD + col] = f2bf(co[r]);
        }
      }
    }
  }
}

// ---------------------------------------------------------------------------
// Kernel 2: output projection. ctx[100352,384](bf16) @ w_out^T + b_out -> f32.
// Per-wave 16x96 tile, 4 column-quarters; A/B frags straight from global
// (ctx + weights are L2-resident). 100352/16 = 6272 row tiles exactly.
// ---------------------------------------------------------------------------
__global__ void out_proj(const bf16_t* __restrict__ ctx,
                         const bf16_t* __restrict__ wout_b,
                         const float* __restrict__ b_out,
                         float* __restrict__ out) {
  const int lane = threadIdx.x & 31;
  const int t  = blockIdx.x * (blockDim.x >> 5) + (threadIdx.x >> 5);
  const int mt = t >> 2;        // 0..6271
  const int nq = t & 3;         // 0..3  (96 columns each)

  v8f acc[6] = {};
  const bf16_t* abase = ctx    + (size_t)mt * 16 * DIM;
  const bf16_t* bbase = wout_b + (size_t)nq * 96 * DIM;
  for (int k0 = 0; k0 < 12; ++k0) {
    v16bf A = load_a_frag(abase + k0 * 32, DIM, lane);
#pragma unroll
    for (int nt = 0; nt < 6; ++nt) {
      v16bf B = load_b_frag(bbase + (size_t)nt * 16 * DIM + k0 * 32, DIM, lane);
      acc[nt] = wmma_bf16(A, B, acc[nt]);
    }
  }
  const int lanelo = lane & 15;
  const int half   = lane >> 4;
#pragma unroll
  for (int nt = 0; nt < 6; ++nt) {
    const int col  = nq * 96 + nt * 16 + lanelo;
    const float bv = b_out[col];
#pragma unroll
    for (int r = 0; r < 8; ++r) {
      const int row = mt * 16 + r + half * 8;
      out[(size_t)row * DIM + col] = acc[nt][r] + bv;
    }
  }
}

// ---------------------------------------------------------------------------
extern "C" void kernel_launch(void* const* d_in, const int* in_sizes, int n_in,
                              void* d_out, int out_size, void* d_ws, size_t ws_size,
                              hipStream_t stream) {
  (void)in_sizes; (void)n_in; (void)out_size; (void)ws_size;
  const float* hidden = (const float*)d_in[0];
  const float* amask  = (const float*)d_in[1];
  const float* wqkv   = (const float*)d_in[2];
  const float* bqkv   = (const float*)d_in[3];
  const float* wout   = (const float*)d_in[4];
  const float* bout   = (const float*)d_in[5];
  const float* table  = (const float*)d_in[6];
  float* out = (float*)d_out;

  char* ws = (char*)d_ws;
  bf16_t* wqkv_b = (bf16_t*)ws;                                   // 3*384*384 bf16
  bf16_t* wout_b = (bf16_t*)(ws + (size_t)3 * DIM * DIM * 2);     // 384*384 bf16
  bf16_t* ctx    = (bf16_t*)(ws + (size_t)4 * DIM * DIM * 2);     // 2048*49*384 bf16

  const int B = 2048;
  const size_t smem = (size_t)(25088 + 6 * 7552) * 2;             // 140800 B
  (void)hipFuncSetAttribute(reinterpret_cast<const void*>(attn_fused),
                            hipFuncAttributeMaxDynamicSharedMemorySize, (int)smem);

  prep_weights<<<(3 * DIM * DIM + 255) / 256, 256, 0, stream>>>(wqkv, wout, wqkv_b, wout_b);
  attn_fused<<<B, 192, smem, stream>>>(hidden, amask, wqkv_b, bqkv, table, ctx);
  out_proj<<<(6272 * 4) / 8, 256, 0, stream>>>(ctx, wout_b, bout, out);
}
